// Model_39676907887389
// MI455X (gfx1250) — matrix operand order repaired and checked by reading, not compiled, not run on hardware
//
#include <hip/hip_runtime.h>

// Problem constants (match reference)
#define B_  2
#define H_  16
#define SQ_ 2048
#define SK_ 2048
#define D_  128

#define KCHUNK 32               // keys processed per main-loop iteration
#define WAVES  8                // waves per block (wave32) -> 256 threads
#define QROWS_PER_BLOCK (WAVES * 16)   // 128 q rows per block

typedef _Float16 half_t;
typedef __attribute__((ext_vector_type(16))) _Float16 v16h;
typedef __attribute__((ext_vector_type(8)))  _Float16 v8h;
typedef __attribute__((ext_vector_type(4)))  _Float16 v4h;
typedef __attribute__((ext_vector_type(8)))  float    v8f;
typedef __attribute__((ext_vector_type(4)))  float    v4f;

__global__ __launch_bounds__(256)
void fa_nosoftmax_kernel(const float* __restrict__ Q,
                         const float* __restrict__ K,
                         const float* __restrict__ V,
                         const int*   __restrict__ M,
                         float*       __restrict__ O)
{
    // LDS: all row strides are multiples of 32B and bank-shifted.
    __shared__ half_t sK [KCHUNK][D_ + 16];       // [key][d]      f16, 288B rows
    __shared__ half_t sVt[D_][KCHUNK + 16];       // [d][key]      f16 (V transposed), 96B rows
    __shared__ half_t sS [WAVES][16][KCHUNK + 16];// wave-private S staging, 96B rows

    const int bh   = blockIdx.x >> 4;     // SQ / 128 = 16 q-blocks per (b,h)
    const int qblk = blockIdx.x & 15;
    const int wave = threadIdx.x >> 5;
    const int lane = threadIdx.x & 31;
    const int q0   = qblk * QROWS_PER_BLOCK + wave * 16;

    const float* Qb = Q + ((size_t)bh * SQ_ + q0) * D_;
    const float* Kb = K + (size_t)bh * SK_ * D_;
    const float* Vb = V + (size_t)bh * SK_ * D_;
    const int*   Mb = M + ((size_t)bh * SQ_ + q0) * (size_t)SK_;
    float*       Ob = O + ((size_t)bh * SQ_ + q0) * D_;

    const int col = lane & 15;   // N-column / A-row selector
    const int hi  = lane >> 4;   // half-wave selector

    // ---- Q fragments: A-matrix layout, 16x32 f16, one per 32-wide d-chunk ----
    // lanes 0-15: K = c2*16..+15 with c2 = lane>>4 ; M = lane&15
    v16h aQ[4];
    #pragma unroll
    for (int c = 0; c < 4; ++c) {
        const float* qrow = Qb + (size_t)col * D_ + c * 32;
        #pragma unroll
        for (int i = 0; i < 8; ++i) {
            aQ[c][i]     = (half_t)qrow[hi * 16 + i];
            aQ[c][8 + i] = (half_t)qrow[hi * 16 + 8 + i];
        }
    }

    v8f zero8 = {};
    v8f acc[8];                  // 16 q-rows x 128 d  (8 n-tiles)
    #pragma unroll
    for (int j = 0; j < 8; ++j) acc[j] = zero8;

    const float scale_keep = 22.62741699796952f;  // sqrt(128) * 1/(1-0.5)

    for (int k0 = 0; k0 < SK_; k0 += KCHUNK) {
        // ---- cooperative staging: K chunk (f16 rows) + V chunk transposed ----
        #pragma unroll
        for (int t = 0; t < 4; ++t) {
            int i  = threadIdx.x + t * 256;   // float4 index within 32x128 tile
            int kk = i >> 5;                  // 32 float4 per row
            int d4 = (i & 31) * 4;
            v4f kv = *(const v4f*)(Kb + (size_t)(k0 + kk) * D_ + d4);
            v4f vv = *(const v4f*)(Vb + (size_t)(k0 + kk) * D_ + d4);
            v4h kh;
            kh[0] = (half_t)kv[0]; kh[1] = (half_t)kv[1];
            kh[2] = (half_t)kv[2]; kh[3] = (half_t)kv[3];
            *(v4h*)&sK[kk][d4] = kh;          // ds_store_b64
            sVt[d4 + 0][kk] = (half_t)vv[0];
            sVt[d4 + 1][kk] = (half_t)vv[1];
            sVt[d4 + 2][kk] = (half_t)vv[2];
            sVt[d4 + 3][kk] = (half_t)vv[3];
        }
        __syncthreads();

        // ---- GEMM1: two 16-key subtiles of S = Q.K^T; scale + dropout; stage ----
        #pragma unroll
        for (int t = 0; t < 2; ++t) {
            v8f s = zero8;
            #pragma unroll
            for (int c = 0; c < 4; ++c) {
                // B-matrix 32x16: lane holds key-column (lane&15), d-rows (lane>>4)*16..+15
                v16h bk = *(const v16h*)&sK[t * 16 + col][c * 32 + hi * 16];
                s = __builtin_amdgcn_wmma_f32_16x16x32_f16(
                        false, aQ[c], false, bk, (short)0, s, false, false);
            }
            const int key = k0 + t * 16 + col;
            #pragma unroll
            for (int r = 0; r < 8; ++r) {
                int   qr   = hi * 8 + r;  // C layout: M = r + 8*(lane>>4)
                int   keep = __builtin_nontemporal_load(&Mb[(size_t)qr * SK_ + key]);
                float val  = keep ? s[r] * scale_keep : 0.0f;
                sS[wave][qr][t * 16 + col] = (half_t)val;
            }
        }

        // ---- reload S in A-matrix layout (lane transpose through LDS) ----
        v8h lo = *(const v8h*)&sS[wave][col][hi * 16];
        v8h hh = *(const v8h*)&sS[wave][col][hi * 16 + 8];
        v16h aS = __builtin_shufflevector(lo, hh,
                     0, 1, 2, 3, 4, 5, 6, 7, 8, 9, 10, 11, 12, 13, 14, 15);

        // ---- GEMM2: O += S.V over this 32-key chunk ----
        #pragma unroll
        for (int j = 0; j < 8; ++j) {
            // B-matrix 32x16: lane holds d-column j*16+(lane&15), k-rows (lane>>4)*16..+15
            v16h bv = *(const v16h*)&sVt[j * 16 + col][hi * 16];
            acc[j] = __builtin_amdgcn_wmma_f32_16x16x32_f16(
                         false, aS, false, bv, (short)0, acc[j], false, false);
        }
        __syncthreads();   // protect sK/sVt before next staging pass
    }

    // ---- epilogue: C layout -> global f32 ----
    #pragma unroll
    for (int j = 0; j < 8; ++j) {
        #pragma unroll
        for (int r = 0; r < 8; ++r) {
            Ob[(size_t)(hi * 8 + r) * D_ + j * 16 + col] = acc[j][r];
        }
    }
}

extern "C" void kernel_launch(void* const* d_in, const int* in_sizes, int n_in,
                              void* d_out, int out_size, void* d_ws, size_t ws_size,
                              hipStream_t stream) {
    (void)in_sizes; (void)n_in; (void)out_size; (void)d_ws; (void)ws_size;
    const float* Q = (const float*)d_in[0];
    const float* K = (const float*)d_in[1];
    const float* V = (const float*)d_in[2];
    const int*   M = (const int*)d_in[3];
    float*       O = (float*)d_out;

    dim3 grid(B_ * H_ * (SQ_ / QROWS_PER_BLOCK));   // 512 blocks
    dim3 block(256);                                 // 8 wave32 waves
    hipLaunchKernelGGL(fa_nosoftmax_kernel, grid, block, 0, stream, Q, K, V, M, O);
}
